// DiacritizerD3_90469191123654
// MI455X (gfx1250) — matrix operand order repaired and checked by reading, not compiled, not run on hardware
//
#include <hip/hip_runtime.h>
#include <hip/hip_bf16.h>

// ---------------------------------------------------------------------------
// Types
// ---------------------------------------------------------------------------
typedef _Float16 half_t;
typedef __attribute__((ext_vector_type(16))) _Float16 v16h;
typedef __attribute__((ext_vector_type(8)))  _Float16 v8h;
typedef __attribute__((ext_vector_type(8)))  float    v8f;
typedef __attribute__((ext_vector_type(4)))  unsigned v4u;
typedef __attribute__((ext_vector_type(8)))  unsigned v8u;

// ---------------------------------------------------------------------------
// Model dims
// ---------------------------------------------------------------------------
#define BB      32
#define TSS     16
#define TWW     16
#define SENT_H  256
#define WORD_H  512
#define CDIM    32
#define WDIM    300
#define WDIM_P  320          // padded to mult of 32
#define DIN     512          // 2*SENT_H
#define DQ      1024         // 2*WORD_H
#define DEC_H   1024
#define DEC_IN  1544
#define DEC_IN_P 1568        // padded to mult of 32
#define CHIN    544          // CDIM + DIN (already mult of 32)
#define NSENT   (BB*TSS)     // 512 sentence rows
#define NTOK    (BB*TSS*TWW) // 8192 char rows
#define DIAC_N  (BB*TSS*TWW*15)

// ---------------------------------------------------------------------------
// Simple WMMA GEMM (one wave per 16x16 tile), used for small/edge shapes:
//   C[M,N] = A[M,K] * B[N,K]^T (+bias[col]); masked store col < Nvalid.
// ---------------------------------------------------------------------------
__global__ __launch_bounds__(32)
void k_gemm_wmma(const half_t* __restrict__ A, const half_t* __restrict__ B,
                 float* __restrict__ C,
                 int M, int N, int K, int lda, int ldb, int ldc,
                 const float* __restrict__ bias, int Nvalid,
                 long long sA, long long sB, long long sC) {
  A += (long long)blockIdx.z * sA;
  B += (long long)blockIdx.z * sB;
  C += (long long)blockIdx.z * sC;
  const int lane = threadIdx.x;
  const int l15  = lane & 15;
  const int kh   = lane >> 4;
  const int mrow = blockIdx.y * 16 + l15;
  const int ncol = blockIdx.x * 16 + l15;

  const half_t* arow = A + (size_t)mrow * lda + kh * 8;
  const half_t* brow = B + (size_t)ncol * ldb + kh * 8;

  v8f acc = {};
  for (int k0 = 0; k0 < K; k0 += 32) {
    v8h alo = *(const v8h*)(arow + k0);
    v8h ahi = *(const v8h*)(arow + k0 + 16);
    v16h a = __builtin_shufflevector(alo, ahi,
        0,1,2,3,4,5,6,7,8,9,10,11,12,13,14,15);
    v8h blo = *(const v8h*)(brow + k0);
    v8h bhi = *(const v8h*)(brow + k0 + 16);
    v16h b = __builtin_shufflevector(blo, bhi,
        0,1,2,3,4,5,6,7,8,9,10,11,12,13,14,15);
    acc = __builtin_amdgcn_wmma_f32_16x16x32_f16(
        false, a, false, b, (short)0, acc, false, false);
  }

  if (ncol < Nvalid) {
    float bv = bias ? bias[ncol] : 0.0f;
#pragma unroll
    for (int r = 0; r < 8; ++r) {
      int rowg = blockIdx.y * 16 + r + kh * 8;
      if (rowg < M) C[(size_t)rowg * ldc + ncol] = acc[r] + bv;
    }
  }
}

// ---------------------------------------------------------------------------
// TDM descriptor helpers (D# group0 / group1, CDNA5 ISA).
// 2D tensor load: tile = 64 rows x 32 f16 elements, row-major, data_size=2B.
// ---------------------------------------------------------------------------
__device__ __forceinline__ v4u tdm_g0(unsigned ldsOff, unsigned long long gaddr) {
  v4u g;
  g.x = 1u;                                            // count=1, user mode
  g.y = ldsOff;                                        // lds_addr
  g.z = (unsigned)gaddr;                               // global_addr[31:0]
  g.w = (unsigned)((gaddr >> 32) & 0x01FFFFFFull)      // global_addr[56:32]
      | (2u << 30);                                    // type=2 ("image")
  return g;
}

__device__ __forceinline__ v8u tdm_g1(unsigned tdim0, unsigned tdim1,
                                      unsigned long long stride0) {
  v8u g;
  g[0] = 0x00010000u;                                  // data_size=1 (2 bytes)
  g[1] = (tdim0 & 0xFFFFu) << 16;                      // tensor_dim0[15:0]
  g[2] = (tdim0 >> 16) | ((tdim1 & 0xFFFFu) << 16);    // tdim0 hi | tdim1 lo
  g[3] = (tdim1 >> 16) | (32u << 16);                  // tdim1 hi | tile_dim0=32
  g[4] = 64u;                                          // tile_dim1=64, tile_dim2=0
  g[5] = (unsigned)stride0;                            // tensor_dim0_stride lo
  g[6] = (unsigned)(stride0 >> 32);                    // stride0 hi | stride1 lo=0
  g[7] = 0u;
  return g;
}

__device__ __forceinline__ void tdm_issue(v4u g0, v8u g1) {
  asm volatile("tensor_load_to_lds %0, %1" :: "s"(g0), "s"(g1) : "memory");
}

// ---------------------------------------------------------------------------
// TDM-fed WMMA GEMM: 4 waves / 128 threads per workgroup, 64x64 C tile,
// 2x2 16x16 tiles per wave. A/B panels (64x32 f16) DMA'd into double-buffered
// LDS via tensor_load_to_lds; chunk c+1 DMA overlaps WMMA on chunk c.
// Requires M%64==0, N%64==0, K%32==0.
// ---------------------------------------------------------------------------
__global__ __launch_bounds__(128)
void k_gemm_tdm(const half_t* __restrict__ A, const half_t* __restrict__ B,
                float* __restrict__ C, int M, int N, int K,
                int lda, int ldb, int ldc, const float* __restrict__ bias) {
  __shared__ __align__(128) half_t Al[2][64 * 32];
  __shared__ __align__(128) half_t Bl[2][64 * 32];

  const int tid  = threadIdx.x;
  const int wave = tid >> 5;
  const int lane = tid & 31;
  const int l15  = lane & 15;
  const int kh   = lane >> 4;
  const int mq   = (wave >> 1) * 32;            // wave's M offset in 64x64 tile
  const int nq   = (wave & 1) * 32;             // wave's N offset
  const long long Mbase = (long long)blockIdx.y * 64;
  const long long Nbase = (long long)blockIdx.x * 64;

  v8f acc00 = {}, acc01 = {}, acc10 = {}, acc11 = {};

  const unsigned ldsA0 = (unsigned)(unsigned long long)(void*)&Al[0][0];
  const unsigned ldsA1 = (unsigned)(unsigned long long)(void*)&Al[1][0];
  const unsigned ldsB0 = (unsigned)(unsigned long long)(void*)&Bl[0][0];
  const unsigned ldsB1 = (unsigned)(unsigned long long)(void*)&Bl[1][0];
  const unsigned long long gA =
      (unsigned long long)A + (unsigned long long)(Mbase * lda) * 2ull;
  const unsigned long long gB =
      (unsigned long long)B + (unsigned long long)(Nbase * ldb) * 2ull;
  const v8u g1A = tdm_g1((unsigned)K, (unsigned)M, (unsigned long long)lda);
  const v8u g1B = tdm_g1((unsigned)K, (unsigned)N, (unsigned long long)ldb);

  const int NC = K / 32;
  if (wave == 0) {
    tdm_issue(tdm_g0(ldsA0, gA), g1A);
    tdm_issue(tdm_g0(ldsB0, gB), g1B);
  }
  for (int c = 0; c < NC; ++c) {
    if (wave == 0) __builtin_amdgcn_s_wait_tensorcnt(0);   // chunk c arrived
    __syncthreads();
    if (wave == 0 && (c + 1) < NC) {                       // prefetch chunk c+1
      unsigned long long ko = (unsigned long long)(c + 1) * 64ull; // 32 elem * 2B
      tdm_issue(tdm_g0((c & 1) ? ldsA0 : ldsA1, gA + ko), g1A);
      tdm_issue(tdm_g0((c & 1) ? ldsB0 : ldsB1, gB + ko), g1B);
    }
    const half_t* Ab = &Al[c & 1][0];
    const half_t* Bb = &Bl[c & 1][0];

    v16h a0, a1, b0, b1;
    {
      const half_t* p = Ab + (mq + l15) * 32 + kh * 8;
      v8h lo = *(const v8h*)p, hi = *(const v8h*)(p + 16);
      a0 = __builtin_shufflevector(lo, hi,
          0,1,2,3,4,5,6,7,8,9,10,11,12,13,14,15);
      p += 16 * 32;
      lo = *(const v8h*)p; hi = *(const v8h*)(p + 16);
      a1 = __builtin_shufflevector(lo, hi,
          0,1,2,3,4,5,6,7,8,9,10,11,12,13,14,15);
    }
    {
      const half_t* p = Bb + (nq + l15) * 32 + kh * 8;
      v8h lo = *(const v8h*)p, hi = *(const v8h*)(p + 16);
      b0 = __builtin_shufflevector(lo, hi,
          0,1,2,3,4,5,6,7,8,9,10,11,12,13,14,15);
      p += 16 * 32;
      lo = *(const v8h*)p; hi = *(const v8h*)(p + 16);
      b1 = __builtin_shufflevector(lo, hi,
          0,1,2,3,4,5,6,7,8,9,10,11,12,13,14,15);
    }
    acc00 = __builtin_amdgcn_wmma_f32_16x16x32_f16(false, a0, false, b0, (short)0, acc00, false, false);
    acc01 = __builtin_amdgcn_wmma_f32_16x16x32_f16(false, a0, false, b1, (short)0, acc01, false, false);
    acc10 = __builtin_amdgcn_wmma_f32_16x16x32_f16(false, a1, false, b0, (short)0, acc10, false, false);
    acc11 = __builtin_amdgcn_wmma_f32_16x16x32_f16(false, a1, false, b1, (short)0, acc11, false, false);
    __syncthreads();
  }

  auto stor = [&](const v8f& acc, long long r0, long long c0) {
    long long col = c0 + l15;
    float bv = bias ? bias[col] : 0.0f;
#pragma unroll
    for (int r = 0; r < 8; ++r) {
      long long row = r0 + r + kh * 8;
      C[row * ldc + col] = acc[r] + bv;
    }
  };
  stor(acc00, Mbase + mq,      Nbase + nq);
  stor(acc01, Mbase + mq,      Nbase + nq + 16);
  stor(acc10, Mbase + mq + 16, Nbase + nq);
  stor(acc11, Mbase + mq + 16, Nbase + nq + 16);
}

// ---------------------------------------------------------------------------
// LSTM gate update: g = pre[:,t,:] + gtmp; i,f,g,o split; state update.
// ---------------------------------------------------------------------------
__global__ void k_lstm_gate(const float* __restrict__ pre, size_t preRS,
                            const float* __restrict__ gtmp,
                            float* __restrict__ cst,
                            half_t* __restrict__ h16,
                            float* __restrict__ out32,
                            half_t* __restrict__ out16, size_t outRS,
                            int N, int H) {
  int idx = blockIdx.x * blockDim.x + threadIdx.x;
  if (idx >= N * H) return;
  int n = idx / H, j = idx % H;
  size_t pb = (size_t)n * preRS;
  size_t gb = (size_t)n * (size_t)(4 * H);
  float gi = pre[pb + j]         + gtmp[gb + j];
  float gf = pre[pb + H + j]     + gtmp[gb + H + j];
  float gg = pre[pb + 2*H + j]   + gtmp[gb + 2*H + j];
  float go = pre[pb + 3*H + j]   + gtmp[gb + 3*H + j];
  float si = 1.0f / (1.0f + __expf(-gi));
  float sf = 1.0f / (1.0f + __expf(-gf));
  float so = 1.0f / (1.0f + __expf(-go));
  float c  = sf * cst[idx] + si * tanhf(gg);
  cst[idx] = c;
  float h  = so * tanhf(c);
  h16[idx] = (half_t)h;
  out16[(size_t)n * outRS + j] = (half_t)h;
  if (out32) out32[(size_t)n * outRS + j] = h;
}

// ---------------------------------------------------------------------------
// f32 -> f16 convert with K padding
// ---------------------------------------------------------------------------
__global__ void k_convert_pad(const float* __restrict__ src, half_t* __restrict__ dst,
                              int rows, int ksrc, int kpad) {
  int idx = blockIdx.x * blockDim.x + threadIdx.x;
  if (idx >= rows * kpad) return;
  int r = idx / kpad, k = idx % kpad;
  dst[idx] = (k < ksrc) ? (half_t)src[(size_t)r * ksrc + k] : (half_t)0.0f;
}

// f32 [srcRows=K, srcCols=N] -> f16 [dstRows>=N, K] transposed (zero-pad rows)
__global__ void k_convert_T(const float* __restrict__ src, half_t* __restrict__ dst,
                            int srcRows, int srcCols, int dstRows) {
  int idx = blockIdx.x * blockDim.x + threadIdx.x;
  if (idx >= dstRows * srcRows) return;
  int n = idx / srcRows, k = idx % srcRows;
  dst[idx] = (n < srcCols) ? (half_t)src[(size_t)k * srcCols + n] : (half_t)0.0f;
}

// ---------------------------------------------------------------------------
// Gather word embeddings -> f16 [NSENT, WDIM_P]
// ---------------------------------------------------------------------------
__global__ void k_gather_wembs(const int* __restrict__ sents,
                               const float* __restrict__ word_embs,
                               half_t* __restrict__ dst) {
  int idx = blockIdx.x * blockDim.x + threadIdx.x;
  if (idx >= NSENT * WDIM_P) return;
  int r = idx / WDIM_P, k = idx % WDIM_P;
  int tok = sents[r];
  dst[idx] = (k < WDIM) ? (half_t)word_embs[(size_t)tok * WDIM + k] : (half_t)0.0f;
}

// ---------------------------------------------------------------------------
// Build char-LSTM input [NTOK, 544]
// ---------------------------------------------------------------------------
__global__ void k_char_in(const int* __restrict__ words,
                          const float* __restrict__ char_table,
                          const half_t* __restrict__ sent_enc16,
                          half_t* __restrict__ dst) {
  int idx = blockIdx.x * blockDim.x + threadIdx.x;
  if (idx >= NTOK * CHIN) return;
  int r = idx / CHIN, d = idx % CHIN;
  int tok = words[r];
  int srow = r >> 4;
  half_t v;
  if (d < CDIM)      v = (half_t)char_table[(size_t)tok * CDIM + d];
  else               v = (tok != 0) ? sent_enc16[(size_t)srow * DIN + (d - CDIM)]
                                    : (half_t)0.0f;
  dst[idx] = v;
}

// ---------------------------------------------------------------------------
// Masked softmax over TS keys -> attn_map region of d_out
// ---------------------------------------------------------------------------
__global__ void k_attn_softmax(const float* __restrict__ scores,
                               const int* __restrict__ words,
                               float* __restrict__ attn_out) {
  int r = blockIdx.x * blockDim.x + threadIdx.x;
  if (r >= NTOK) return;
  int b = r >> 8;
  int i = (r >> 4) & 15;
  float vals[TSS];
  float mx = -1e30f;
#pragma unroll
  for (int j = 0; j < TSS; ++j) {
    bool kv = false;
    for (int w = 0; w < TWW; ++w) kv = kv || (words[b*256 + j*16 + w] != 0);
    bool valid = kv && (j != i);
    float s = valid ? scores[(size_t)r * TSS + j] : -1e9f;
    vals[j] = s;
    mx = fmaxf(mx, s);
  }
  float sum = 0.0f;
#pragma unroll
  for (int j = 0; j < TSS; ++j) { vals[j] = __expf(vals[j] - mx); sum += vals[j]; }
  float inv = 1.0f / sum;
#pragma unroll
  for (int j = 0; j < TSS; ++j) attn_out[(size_t)r * TSS + j] = vals[j] * inv;
}

// ---------------------------------------------------------------------------
// attn_enc = attn_map @ sent_enc (K=16) -> f16 into final16 cols [0,512)
// ---------------------------------------------------------------------------
__global__ void k_attn_enc(const float* __restrict__ attn,
                           const float* __restrict__ sent_enc32,
                           half_t* __restrict__ final16) {
  int idx = blockIdx.x * blockDim.x + threadIdx.x;
  if (idx >= NTOK * DIN) return;
  int r = idx >> 9, d = idx & 511;
  int b = r >> 8;
  const float* arow = attn + (size_t)r * TSS;
  const float* se   = sent_enc32 + (size_t)b * TSS * DIN + d;
  float s = 0.0f;
#pragma unroll
  for (int j = 0; j < TSS; ++j) s += arow[j] * se[(size_t)j * DIN];
  final16[(size_t)r * DEC_IN_P + d] = (half_t)s;
}

// Remaining decoder-input cols: char_enc (512..1535), labels (1536..1543), pad 0
__global__ void k_final_rest(const half_t* __restrict__ char_enc16,
                             const int* __restrict__ labels,
                             half_t* __restrict__ final16) {
  const int span = DEC_IN_P - DIN;
  int idx = blockIdx.x * blockDim.x + threadIdx.x;
  if (idx >= NTOK * span) return;
  int r = idx / span, d = DIN + idx % span;
  half_t v;
  if (d < DIN + DQ)         v = char_enc16[(size_t)r * DQ + (d - DIN)];
  else if (d < DEC_IN)      v = (half_t)(float)labels[(size_t)r * 8 + (d - DIN - DQ)];
  else                      v = (half_t)0.0f;
  final16[(size_t)r * DEC_IN_P + d] = v;
}

// ---------------------------------------------------------------------------
// Host orchestration
// ---------------------------------------------------------------------------
extern "C" void kernel_launch(void* const* d_in, const int* in_sizes, int n_in,
                              void* d_out, int out_size, void* d_ws, size_t ws_size,
                              hipStream_t stream) {
  (void)in_sizes; (void)n_in; (void)out_size; (void)ws_size;

  const int*   sents      = (const int*)  d_in[0];
  const int*   words      = (const int*)  d_in[1];
  const int*   labels     = (const int*)  d_in[2];
  const float* word_embs  = (const float*)d_in[3];
  const float* char_table = (const float*)d_in[4];
  const float* sent_Wx0   = (const float*)d_in[5];
  const float* sent_Wh0   = (const float*)d_in[6];
  const float* sent_b0    = (const float*)d_in[7];
  const float* sent_Wx1   = (const float*)d_in[8];
  const float* sent_Wh1   = (const float*)d_in[9];
  const float* sent_b1    = (const float*)d_in[10];
  const float* word_Wx0   = (const float*)d_in[11];
  const float* word_Wh0   = (const float*)d_in[12];
  const float* word_b0    = (const float*)d_in[13];
  const float* word_Wx1   = (const float*)d_in[14];
  const float* word_Wh1   = (const float*)d_in[15];
  const float* word_b1    = (const float*)d_in[16];
  const float* attn_Wq    = (const float*)d_in[17];
  const float* dec_Wx     = (const float*)d_in[18];
  const float* dec_Wh     = (const float*)d_in[19];
  const float* dec_b      = (const float*)d_in[20];
  const float* cls_W      = (const float*)d_in[21];
  const float* cls_b      = (const float*)d_in[22];
  float* out = (float*)d_out;

  size_t off = 0;
  auto alloc = [&](size_t bytes) -> char* {
    char* p = (char*)d_ws + off;
    off = (off + bytes + 255) & ~(size_t)255;
    return p;
  };
  half_t* wSWx0 = (half_t*)alloc((size_t)2*1024*WDIM_P*2);
  half_t* wSWh0 = (half_t*)alloc((size_t)2*1024*SENT_H*2);
  half_t* wSWx1 = (half_t*)alloc((size_t)2*1024*DIN*2);
  half_t* wSWh1 = (half_t*)alloc((size_t)2*1024*SENT_H*2);
  half_t* wWWx0 = (half_t*)alloc((size_t)2*2048*CHIN*2);
  half_t* wWWh0 = (half_t*)alloc((size_t)2*2048*WORD_H*2);
  half_t* wWWx1 = (half_t*)alloc((size_t)2*2048*DQ*2);
  half_t* wWWh1 = (half_t*)alloc((size_t)2*2048*WORD_H*2);
  half_t* wAttnT= (half_t*)alloc((size_t)DIN*DQ*2);
  half_t* wDWx  = (half_t*)alloc((size_t)4096*DEC_IN_P*2);
  half_t* wDWh  = (half_t*)alloc((size_t)4096*DEC_H*2);
  half_t* wClsT = (half_t*)alloc((size_t)16*DEC_H*2);
  half_t* wembs16   = (half_t*)alloc((size_t)NSENT*WDIM_P*2);
  float*  sent_pre  = (float*) alloc((size_t)NSENT*1024*4);
  half_t* sentL0_16 = (half_t*)alloc((size_t)NSENT*DIN*2);
  half_t* sentE16   = (half_t*)alloc((size_t)NSENT*DIN*2);
  float*  sentE32   = (float*) alloc((size_t)NSENT*DIN*4);
  half_t* char_in16 = (half_t*)alloc((size_t)NTOK*CHIN*2);
  float*  big_pre   = (float*) alloc((size_t)NTOK*4096*4);
  half_t* charL0_16 = (half_t*)alloc((size_t)NTOK*DQ*2);
  half_t* charE16   = (half_t*)alloc((size_t)NTOK*DQ*2);
  float*  q32       = (float*) alloc((size_t)NTOK*DIN*4);
  half_t* q16       = (half_t*)alloc((size_t)NTOK*DIN*2);
  float*  scores    = (float*) alloc((size_t)NTOK*TSS*4);
  half_t* final16   = (half_t*)alloc((size_t)NTOK*DEC_IN_P*2);
  half_t* decOut16  = (half_t*)alloc((size_t)NTOK*DEC_H*2);
  float*  gtmp      = (float*) alloc((size_t)512*2048*4);
  half_t* h16buf    = (half_t*)alloc((size_t)512*512*2);
  float*  cbuf      = (float*) alloc((size_t)512*512*4);

  auto gemm = [&](const half_t* A, const half_t* Bm, float* C,
                  int M, int N, int K, int lda, int ldb, int ldc,
                  const float* bias, int Nvalid, int batches,
                  long long sA, long long sB, long long sC) {
    if (batches == 1 && (M % 64) == 0 && (N % 64) == 0 && Nvalid == N) {
      dim3 g(N/64, M/64);
      k_gemm_tdm<<<g, 128, 0, stream>>>(A, Bm, C, M, N, K, lda, ldb, ldc, bias);
    } else {
      dim3 g(N/16, M/16, batches);
      k_gemm_wmma<<<g, 32, 0, stream>>>(A, Bm, C, M, N, K, lda, ldb, ldc,
                                        bias, Nvalid, sA, sB, sC);
    }
  };
  auto conv = [&](const float* s, half_t* d, int rows, int ks, int kp) {
    int tot = rows * kp;
    k_convert_pad<<<(tot+255)/256, 256, 0, stream>>>(s, d, rows, ks, kp);
  };
  auto scan = [&](const float* pre, int T, int H, int H4, int N,
                  const half_t* Wh, half_t* out16, float* out32,
                  int outLD, bool reverse) {
    hipMemsetAsync(h16buf, 0, (size_t)N*H*sizeof(half_t), stream);
    hipMemsetAsync(cbuf,   0, (size_t)N*H*sizeof(float),  stream);
    for (int s = 0; s < T; ++s) {
      int t = reverse ? (T-1-s) : s;
      gemm(h16buf, Wh, gtmp, N, H4, H, H, H, H4, nullptr, H4, 1, 0, 0, 0);
      int tot = N * H;
      k_lstm_gate<<<(tot+255)/256, 256, 0, stream>>>(
          pre + (size_t)t*H4, (size_t)T*H4, gtmp, cbuf, h16buf,
          out32 ? out32 + (size_t)t*outLD : nullptr,
          out16 + (size_t)t*outLD, (size_t)T*outLD, N, H);
    }
  };

  // ---- 0) weight conversion to f16 (padded) ----
  conv(sent_Wx0, wSWx0, 2*1024, WDIM,   WDIM_P);
  conv(sent_Wh0, wSWh0, 2*1024, SENT_H, SENT_H);
  conv(sent_Wx1, wSWx1, 2*1024, DIN,    DIN);
  conv(sent_Wh1, wSWh1, 2*1024, SENT_H, SENT_H);
  conv(word_Wx0, wWWx0, 2*2048, CHIN,   CHIN);
  conv(word_Wh0, wWWh0, 2*2048, WORD_H, WORD_H);
  conv(word_Wx1, wWWx1, 2*2048, DQ,     DQ);
  conv(word_Wh1, wWWh1, 2*2048, WORD_H, WORD_H);
  conv(dec_Wx,   wDWx,  4096,   DEC_IN, DEC_IN_P);
  conv(dec_Wh,   wDWh,  4096,   DEC_H,  DEC_H);
  { int tot = DIN*DQ;
    k_convert_T<<<(tot+255)/256,256,0,stream>>>(attn_Wq, wAttnT, DQ, DIN, DIN); }
  { int tot = 16*DEC_H;
    k_convert_T<<<(tot+255)/256,256,0,stream>>>(cls_W, wClsT, DEC_H, 15, 16); }

  // ---- 1) sentence encoder ----
  { int tot = NSENT*WDIM_P;
    k_gather_wembs<<<(tot+255)/256,256,0,stream>>>(sents, word_embs, wembs16); }
  for (int d = 0; d < 2; ++d) {   // layer 0
    gemm(wembs16, wSWx0 + (size_t)d*1024*WDIM_P, sent_pre,
         NSENT, 1024, WDIM_P, WDIM_P, WDIM_P, 1024, sent_b0 + d*1024, 1024, 1,0,0,0);
    scan(sent_pre, TSS, SENT_H, 1024, BB, wSWh0 + (size_t)d*1024*SENT_H,
         sentL0_16 + d*SENT_H, nullptr, DIN, d == 1);
  }
  for (int d = 0; d < 2; ++d) {   // layer 1
    gemm(sentL0_16, wSWx1 + (size_t)d*1024*DIN, sent_pre,
         NSENT, 1024, DIN, DIN, DIN, 1024, sent_b1 + d*1024, 1024, 1,0,0,0);
    scan(sent_pre, TSS, SENT_H, 1024, BB, wSWh1 + (size_t)d*1024*SENT_H,
         sentE16 + d*SENT_H, sentE32 + d*SENT_H, DIN, d == 1);
  }

  // ---- 2) char encoder ----
  { int tot = NTOK*CHIN;
    k_char_in<<<(tot+255)/256,256,0,stream>>>(words, char_table, sentE16, char_in16); }
  for (int d = 0; d < 2; ++d) {   // layer 0
    gemm(char_in16, wWWx0 + (size_t)d*2048*CHIN, big_pre,
         NTOK, 2048, CHIN, CHIN, CHIN, 2048, word_b0 + d*2048, 2048, 1,0,0,0);
    scan(big_pre, TWW, WORD_H, 2048, NSENT, wWWh0 + (size_t)d*2048*WORD_H,
         charL0_16 + d*WORD_H, nullptr, DQ, d == 1);
  }
  for (int d = 0; d < 2; ++d) {   // layer 1
    gemm(charL0_16, wWWx1 + (size_t)d*2048*DQ, big_pre,
         NTOK, 2048, DQ, DQ, DQ, 2048, word_b1 + d*2048, 2048, 1,0,0,0);
    scan(big_pre, TWW, WORD_H, 2048, NSENT, wWWh1 + (size_t)d*2048*WORD_H,
         charE16 + d*WORD_H, nullptr, DQ, d == 1);
  }

  // ---- 3) attention ----
  gemm(charE16, wAttnT, q32, NTOK, DIN, DQ, DQ, DQ, DIN, nullptr, DIN, 1,0,0,0);
  conv(q32, q16, NTOK, DIN, DIN);
  gemm(q16, sentE16, scores, 256, TSS, DIN, DIN, DIN, TSS, nullptr, TSS,
       BB, (long long)256*DIN, (long long)TSS*DIN, (long long)256*TSS);
  float* attn_out = out + DIAC_N;
  k_attn_softmax<<<(NTOK+255)/256,256,0,stream>>>(scores, words, attn_out);
  { int tot = NTOK*DIN;
    k_attn_enc<<<(tot+255)/256,256,0,stream>>>(attn_out, sentE32, final16); }
  { int tot = NTOK*(DEC_IN_P-DIN);
    k_final_rest<<<(tot+255)/256,256,0,stream>>>(charE16, labels, final16); }

  // ---- 4) decoder LSTM ----
  gemm(final16, wDWx, big_pre, NTOK, 4096, DEC_IN_P, DEC_IN_P, DEC_IN_P, 4096,
       dec_b, 4096, 1,0,0,0);
  scan(big_pre, TSS*TWW, DEC_H, 4096, BB, wDWh, decOut16, nullptr, DEC_H, false);

  // ---- 5) classifier ----
  gemm(decOut16, wClsT, out, NTOK, 16, DEC_H, DEC_H, DEC_H, 15, cls_b, 15, 1,0,0,0);
}
